// RawSingleAttention_8263517077812
// MI455X (gfx1250) — matrix operand order repaired and checked by reading, not compiled, not run on hardware
//
#include <hip/hip_runtime.h>

#define NQc 2048
#define NKc 2048
#define DDc 1024
#define BM 32
#define BN 32
#define NWAVES 16
#define BLOCK (NWAVES * 32)

typedef __attribute__((ext_vector_type(16))) __bf16   bf16x16;
typedef __attribute__((ext_vector_type(8)))  float    f32x8;
typedef __attribute__((ext_vector_type(4)))  unsigned u32x4;

union FragBF { u32x4 u[2]; bf16x16 v; };

__device__ __forceinline__ unsigned short f2bf(float x) {
    unsigned u = __builtin_bit_cast(unsigned, x);
    u += 0x7FFFu + ((u >> 16) & 1u);      // round-to-nearest-even
    return (unsigned short)(u >> 16);
}

__global__ __launch_bounds__(BLOCK)
void fa_causal_wmma_kernel(const float* __restrict__ Q,
                           const float* __restrict__ K,
                           const float* __restrict__ V,
                           const float* __restrict__ Msk,
                           float* __restrict__ O)
{
    __shared__ __align__(16) unsigned short Ks[BN][DDc];     // 64 KB (Q staged here first)
    __shared__ __align__(16) unsigned short VsT[DDc][BN];    // 64 KB, V transposed [col][row]
    __shared__ __align__(16) float          Ssum[2][16][BN]; // 4 KB
    __shared__ __align__(16) unsigned short Ps[2][16][BN];   // 2 KB

    const int tid  = threadIdx.x;
    const int wave = tid >> 5;
    const int lane = tid & 31;
    const int mi   = wave & 1;       // which 16-row M tile
    const int ni   = wave >> 1;      // 0..7 -> 128-col DV slice / 128-wide k slice
    const int hgrp = lane >> 4;      // lane half-group (0/1)
    const int l16  = lane & 15;

    const int qi0 = blockIdx.x * BM;
    const int b   = blockIdx.y;

    const float* Qb = Q   + (size_t)b * NQc * DDc;
    const float* Kb = K   + (size_t)b * NKc * DDc;
    const float* Vb = V   + (size_t)b * NKc * DDc;
    const float* Mb = Msk + (size_t)b * NQc * (size_t)NKc;
    float*       Ob = O   + (size_t)b * NQc * DDc;

    // ---------------- stage Q (fp32 -> bf16) into Ks, capture A fragments ----
    for (int idx = tid; idx < BM * (DDc / 4); idx += BLOCK) {
        int row = idx >> 8;                   // DDc/4 == 256
        int c   = (idx & 255) << 2;
        const float4 q4 = *(const float4*)(Qb + (size_t)(qi0 + row) * DDc + c);
        unsigned p0 = (unsigned)f2bf(q4.x) | ((unsigned)f2bf(q4.y) << 16);
        unsigned p1 = (unsigned)f2bf(q4.z) | ((unsigned)f2bf(q4.w) << 16);
        *(unsigned*)&Ks[row][c]     = p0;
        *(unsigned*)&Ks[row][c + 2] = p1;
    }
    __syncthreads();

    FragBF qf[4];                             // this wave's k window: [ni*128, ni*128+128)
#pragma unroll
    for (int kk = 0; kk < 4; ++kk) {
        int kb = ni * 128 + kk * 32;
        qf[kk].u[0] = *(const u32x4*)&Ks[mi * 16 + l16][kb + 16 * hgrp];
        qf[kk].u[1] = *(const u32x4*)&Ks[mi * 16 + l16][kb + 16 * hgrp + 8];
    }
    __syncthreads();

    f32x8 acc[8] = {};
    float mrow[8], lrow[8];
#pragma unroll
    for (int r = 0; r < 8; ++r) { mrow[r] = -1e30f; lrow[r] = 0.f; }

    const float LOG2E = 1.4426950408889634f;
    const int ntiles = (qi0 >> 5) + 1;        // causal: only j0 <= qi0

    for (int t = 0; t < ntiles; ++t) {
        const int  j0   = t * BN;
        const bool diag = (j0 + BN > qi0);    // only the last tile touches the mask

        // ---- zero S reduction buffer + stage K (bf16) and V^T (bf16) -------
        for (int i = tid; i < 2 * 16 * BN; i += BLOCK) ((float*)Ssum)[i] = 0.f;
        for (int idx = tid; idx < BN * (DDc / 4); idx += BLOCK) {
            int row = idx >> 8;
            int c   = (idx & 255) << 2;
            const float4 k4 = *(const float4*)(Kb + (size_t)(j0 + row) * DDc + c);
            unsigned kp0 = (unsigned)f2bf(k4.x) | ((unsigned)f2bf(k4.y) << 16);
            unsigned kp1 = (unsigned)f2bf(k4.z) | ((unsigned)f2bf(k4.w) << 16);
            *(unsigned*)&Ks[row][c]     = kp0;
            *(unsigned*)&Ks[row][c + 2] = kp1;
            const float4 v4 = *(const float4*)(Vb + (size_t)(j0 + row) * DDc + c);
            VsT[c + 0][row] = f2bf(v4.x);
            VsT[c + 1][row] = f2bf(v4.y);
            VsT[c + 2][row] = f2bf(v4.z);
            VsT[c + 3][row] = f2bf(v4.w);
        }
        if (t + 1 < ntiles) {                 // hint next tile into cache
            const char* pk = (const char*)(Kb + (size_t)(j0 + BN) * DDc);
            const char* pv = (const char*)(Vb + (size_t)(j0 + BN) * DDc);
            __builtin_prefetch(pk + (size_t)tid * 256, 0, 1);
            __builtin_prefetch(pv + (size_t)tid * 256, 0, 1);
        }
        __syncthreads();

        // ---- S = Q K^T : partial over this wave's 128-wide k slice ---------
        f32x8 sp0 = {}, sp1 = {};
#pragma unroll
        for (int kk = 0; kk < 4; ++kk) {
            int kb = ni * 128 + kk * 32 + 16 * hgrp;
            FragBF b0, b1;
            b0.u[0] = *(const u32x4*)&Ks[l16][kb];
            b0.u[1] = *(const u32x4*)&Ks[l16][kb + 8];
            b1.u[0] = *(const u32x4*)&Ks[16 + l16][kb];
            b1.u[1] = *(const u32x4*)&Ks[16 + l16][kb + 8];
            sp0 = __builtin_amdgcn_wmma_f32_16x16x32_bf16(false, qf[kk].v, false, b0.v,
                                                          (short)0, sp0, false, false);
            sp1 = __builtin_amdgcn_wmma_f32_16x16x32_bf16(false, qf[kk].v, false, b1.v,
                                                          (short)0, sp1, false, false);
        }
#pragma unroll
        for (int r = 0; r < 8; ++r) {         // C layout: VGPR r -> row r / r+8
            int m = r + 8 * hgrp;
            atomicAdd(&Ssum[mi][m][l16],      sp0[r]);
            atomicAdd(&Ssum[mi][m][16 + l16], sp1[r]);
        }
        __syncthreads();

        // ---- scale + mask + online softmax ---------------------------------
        float pr0[8], pr1[8];
#pragma unroll
        for (int r = 0; r < 8; ++r) {
            int m = r + 8 * hgrp;
            float s0 = Ssum[mi][m][l16];
            float s1 = Ssum[mi][m][16 + l16];
            if (diag) {
                const float* mp = Mb + (size_t)(qi0 + mi * 16 + m) * NKc + j0;
                float mv0 = mp[l16];
                float mv1 = mp[16 + l16];
                s0 = (mv0 < -1.0f) ? mv0 : s0 * 0.03125f;   // 1/sqrt(1024)
                s1 = (mv1 < -1.0f) ? mv1 : s1 * 0.03125f;
            } else { s0 *= 0.03125f; s1 *= 0.03125f; }

            float rm = fmaxf(s0, s1);                       // 16-lane row reduce
            rm = fmaxf(rm, __shfl_xor(rm, 1, 32));
            rm = fmaxf(rm, __shfl_xor(rm, 2, 32));
            rm = fmaxf(rm, __shfl_xor(rm, 4, 32));
            rm = fmaxf(rm, __shfl_xor(rm, 8, 32));
            float mnew  = fmaxf(mrow[r], rm);
            float alpha = __builtin_amdgcn_exp2f((mrow[r] - mnew) * LOG2E);
            float p0    = __builtin_amdgcn_exp2f((s0 - mnew) * LOG2E);
            float p1    = __builtin_amdgcn_exp2f((s1 - mnew) * LOG2E);
            float rs = p0 + p1;
            rs += __shfl_xor(rs, 1, 32);
            rs += __shfl_xor(rs, 2, 32);
            rs += __shfl_xor(rs, 4, 32);
            rs += __shfl_xor(rs, 8, 32);
            lrow[r] = lrow[r] * alpha + rs;
            mrow[r] = mnew;
            pr0[r] = p0; pr1[r] = p1;
#pragma unroll
            for (int nt = 0; nt < 8; ++nt) acc[nt][r] *= alpha;
        }
        if (ni == 0) {                         // one wave per M-tile publishes P
#pragma unroll
            for (int r = 0; r < 8; ++r) {
                int m = r + 8 * hgrp;
                Ps[mi][m][l16]      = f2bf(pr0[r]);
                Ps[mi][m][16 + l16] = f2bf(pr1[r]);
            }
        }
        __syncthreads();

        // ---- O += P V : 8 WMMAs over this wave's 128 DV columns ------------
        FragBF pf;
        pf.u[0] = *(const u32x4*)&Ps[mi][l16][16 * hgrp];
        pf.u[1] = *(const u32x4*)&Ps[mi][l16][16 * hgrp + 8];
#pragma unroll
        for (int nt = 0; nt < 8; ++nt) {
            int col = ni * 128 + nt * 16 + l16;
            FragBF vf;
            vf.u[0] = *(const u32x4*)&VsT[col][16 * hgrp];
            vf.u[1] = *(const u32x4*)&VsT[col][16 * hgrp + 8];
            acc[nt] = __builtin_amdgcn_wmma_f32_16x16x32_bf16(false, pf.v, false, vf.v,
                                                              (short)0, acc[nt], false, false);
        }
        __syncthreads();                       // protect LDS before next staging
    }

    // ---- epilogue: O = acc / l  (l==0 -> 0, mirrors NaN->0) ----------------
#pragma unroll
    for (int r = 0; r < 8; ++r) {
        int   m   = r + 8 * hgrp;
        float inv = lrow[r] > 0.f ? 1.0f / lrow[r] : 0.f;
        float* orow = Ob + (size_t)(qi0 + mi * 16 + m) * DDc + ni * 128 + l16;
#pragma unroll
        for (int nt = 0; nt < 8; ++nt)
            orow[nt * 16] = acc[nt][r] * inv;
    }
}

extern "C" void kernel_launch(void* const* d_in, const int* in_sizes, int n_in,
                              void* d_out, int out_size, void* d_ws, size_t ws_size,
                              hipStream_t stream) {
    const float* Q = (const float*)d_in[0];
    const float* K = (const float*)d_in[1];
    const float* V = (const float*)d_in[2];
    const float* M = (const float*)d_in[3];
    float*       O = (float*)d_out;
    const int Bsz = in_sizes[0] / (NQc * DDc);
    dim3 grid(NQc / BM, Bsz);
    fa_causal_wmma_kernel<<<grid, BLOCK, 0, stream>>>(Q, K, V, M, O);
}
